// MultiHeadLinearAttention2D_21157008900665
// MI455X (gfx1250) — matrix operand order repaired and checked by reading, not compiled, not run on hardware
//
#include <hip/hip_runtime.h>
#include <hip/hip_bf16.h>

// Multi-head linear attention, fused into 2 passes:
//  K1: k,v projections (WMMA f16) -> KV[b,h,32,32], k_sum[b,h,32] (atomic accum)
//  K2: q projection (WMMA f16) -> out = (KV^T q) / max(q.k_sum, 1e-6)

#define HEADS   8
#define BATCH   16
#define CIN     256
#define COUT    256
#define MSPAT   4096        // 64*64
#define MT      128         // spatial columns per workgroup
#define KCH     32          // channels per WMMA K-step
#define NKC     (CIN / KCH) // 8
#define DH      (COUT / HEADS) // 32
#define NRT     (COUT / 16)    // 16 row tiles of the projection output

// strides (in halves): weight/x staging rows padded to 36 (8B aligned rows),
// k/v tiles 132, q^T 264 (16B aligned rows), KV^T 34.
#define XS_S   36
#define W_S    36
#define KT_S   132
#define QT_S   264
#define KVT_S  34

typedef __attribute__((ext_vector_type(16))) _Float16     v16h;
typedef __attribute__((ext_vector_type(2)))  _Float16     h2;
typedef __attribute__((ext_vector_type(8)))  float        v8f;
typedef __attribute__((ext_vector_type(8)))  unsigned int v8u;
typedef __attribute__((ext_vector_type(2)))  unsigned int v2u;
typedef __attribute__((ext_vector_type(4)))  unsigned int v4u;

static __device__ __forceinline__ v16h as_h16(v8u u) {
  return __builtin_bit_cast(v16h, u);
}
static __device__ __forceinline__ unsigned int pk2(float a, float b) {
  // v_cvt_pk_f16_f32 (RTZ variant): two f32 -> packed f16x2 in one dword
  return __builtin_bit_cast(unsigned int, __builtin_amdgcn_cvt_pkrtz(a, b));
}
static __device__ __forceinline__ v8f wmma_f16(v16h a, v16h b, v8f c) {
  // emits v_wmma_f32_16x16x32_f16
  return __builtin_amdgcn_wmma_f32_16x16x32_f16(false, a, false, b, (short)0, c,
                                                false, false);
}
static __device__ __forceinline__ v8f v8f_zero() {
  v8f z;
#pragma unroll
  for (int e = 0; e < 8; ++e) z[e] = 0.0f;
  return z;
}
static __device__ __forceinline__ float feat_scale(float x) {
  // (elu(x)+1) * 1/sqrt(dh);  elu(x)+1 == x+1 (x>0) else exp(x)
  float e = (x > 0.0f) ? (x + 1.0f) : __expf(x);
  return e * 0.17677669529663687f; // 1/sqrt(32)
}

// ---------------------------------------------------------------------------
// Kernel 1: KV and k_sum accumulation
// LDS: phase A  : xs[128][36] | wks[256][36] | wvs[256][36]  f16  (46KB)
//      phase B/C: kt[256][132] | vt[256][132]                f16  (132KB)
// regions unioned (phase A fully consumed before phase B writes)
// ---------------------------------------------------------------------------
__global__ void __launch_bounds__(256)
mhla_kv_kernel(const float* __restrict__ x, const float* __restrict__ wk,
               const float* __restrict__ wv, float* __restrict__ kv_ws,
               float* __restrict__ ksum_ws) {
  __shared__ __align__(16) char smem[135168];
  const int b    = blockIdx.y;
  const int m0   = blockIdx.x * MT;
  const int tid  = threadIdx.x;
  const int lane = tid & 31;
  const int wave = tid >> 5;   // 0..7 -> 16-col subtile / head id
  const int ln15 = lane & 15;
  const int lhi  = lane >> 4;  // 0/1: which K-half this lane carries

  _Float16* xs  = (_Float16*)smem;        // [col][chan]  stride XS_S
  _Float16* wks = xs + MT * XS_S;         // [row][chan]  stride W_S
  _Float16* wvs = wks + COUT * W_S;
  _Float16* kt  = (_Float16*)smem;        // [row][col]   stride KT_S
  _Float16* vt  = kt + COUT * KT_S;

  v8f acc_k[NRT], acc_v[NRT];
#pragma unroll
  for (int rt = 0; rt < NRT; ++rt) { acc_k[rt] = v8f_zero(); acc_v[rt] = v8f_zero(); }

  const unsigned int* xsu  = (const unsigned int*)xs;
  const unsigned int* wksu = (const unsigned int*)wks;
  const unsigned int* wvsu = (const unsigned int*)wvs;

  for (int kc = 0; kc < NKC; ++kc) {
    const int c0 = kc * KCH;
    // stage x chunk [32 chan x 128 col] -> f16 transposed; float4 global loads
#pragma unroll
    for (int i = 0; i < (KCH * MT) / (256 * 4); ++i) {   // 4
      const int e   = (tid + i * 256) * 4;
      const int kk  = e >> 7;
      const int col = e & (MT - 1);
      const float4 f =
          *(const float4*)&x[((size_t)b * CIN + c0 + kk) * MSPAT + m0 + col];
      xs[(col + 0) * XS_S + kk] = (_Float16)f.x;
      xs[(col + 1) * XS_S + kk] = (_Float16)f.y;
      xs[(col + 2) * XS_S + kk] = (_Float16)f.z;
      xs[(col + 3) * XS_S + kk] = (_Float16)f.w;
    }
    // stage Wk / Wv K-slices [256 rows x 32 chan]; float4 loads, b64 LDS stores
#pragma unroll
    for (int i = 0; i < (COUT * KCH) / (256 * 4); ++i) { // 8
      const int e   = (tid + i * 256) * 4;
      const int row = e >> 5;
      const int kk  = e & 31;            // multiple of 4 -> 8B-aligned store
      const float4 fk = *(const float4*)&wk[row * CIN + c0 + kk];
      const float4 fv = *(const float4*)&wv[row * CIN + c0 + kk];
      v2u pk = {pk2(fk.x, fk.y), pk2(fk.z, fk.w)};
      v2u pv = {pk2(fv.x, fv.y), pk2(fv.z, fv.w)};
      *(v2u*)&wks[row * W_S + kk] = pk;
      *(v2u*)&wvs[row * W_S + kk] = pv;
    }
    if (kc + 1 < NKC) {
      __builtin_prefetch(&x[((size_t)b * CIN + c0 + KCH) * MSPAT + m0 + lane * 4], 0, 1);
    }
    __syncthreads();

    // B fragment: x[K=32 chan][N=16 cols of this wave]
    {
      const int col = wave * 16 + ln15;
      const int kb2 = (lhi * 8) >> 1;  // dword offset of this lane's K-group (table order)
      v8u bu;
#pragma unroll
      for (int j = 0; j < 8; ++j) bu[j] = xsu[col * (XS_S / 2) + kb2 + j + (j >> 2) * 4];
      const v16h Bf = as_h16(bu);

      const int akb2 = (lhi * 8) >> 1;
#pragma unroll
      for (int rt = 0; rt < NRT; ++rt) {
        const int rbase = (rt * 16 + ln15) * (W_S / 2);
        v8u au;
#pragma unroll
        for (int j = 0; j < 4; ++j) au[j] = wksu[rbase + akb2 + j];
#pragma unroll
        for (int j = 0; j < 4; ++j) au[4 + j] = wksu[rbase + 8 + akb2 + j];
        acc_k[rt] = wmma_f16(as_h16(au), Bf, acc_k[rt]);
        v8u av;
#pragma unroll
        for (int j = 0; j < 4; ++j) av[j] = wvsu[rbase + akb2 + j];
#pragma unroll
        for (int j = 0; j < 4; ++j) av[4 + j] = wvsu[rbase + 8 + akb2 + j];
        acc_v[rt] = wmma_f16(as_h16(av), Bf, acc_v[rt]);
      }
    }
    __syncthreads();
  }

  // Phase B: activation, spill k/v tiles to LDS
  {
    const int colw = wave * 16 + ln15;
#pragma unroll
    for (int rt = 0; rt < NRT; ++rt) {
#pragma unroll
      for (int j = 0; j < 8; ++j) {
        const int crow = rt * 16 + j + lhi * 8;  // D-frag row mapping
        kt[crow * KT_S + colw] = (_Float16)feat_scale(acc_k[rt][j]);
        vt[crow * KT_S + colw] = (_Float16)acc_v[rt][j];
      }
    }
  }
  __syncthreads();

  // k_sum over this tile's 128 columns (one row per thread, packed-h2 reads)
  {
    const h2* ktp = (const h2*)kt;
    float s = 0.0f;
#pragma unroll 8
    for (int c2 = 0; c2 < MT / 2; ++c2) {
      h2 p = ktp[tid * (KT_S / 2) + c2];
      s += (float)p[0] + (float)p[1];
    }
    atomicAdd(&ksum_ws[b * COUT + tid], s);
  }

  // Phase C: KV[h] += k[h] * v[h]^T  (A=k 16x32-of-m, B=v^T 32x16)
  {
    const unsigned int* ktu = (const unsigned int*)kt;
    const unsigned int* vtu = (const unsigned int*)vt;
    const int h = wave;
    v8f kv[2][2];
#pragma unroll
    for (int ct = 0; ct < 2; ++ct)
#pragma unroll
      for (int dt = 0; dt < 2; ++dt) kv[ct][dt] = v8f_zero();

#pragma unroll
    for (int mc = 0; mc < MT / KCH; ++mc) {  // 4 K-steps over m
      const int mA2 = (mc * KCH + lhi * 8) >> 1;
      const int mB2 = (mc * KCH + lhi * 8) >> 1;
      v16h Af[2], Bf[2];
#pragma unroll
      for (int ct = 0; ct < 2; ++ct) {
        const int row = (h * DH + ct * 16 + ln15) * (KT_S / 2);
        v8u au;
#pragma unroll
        for (int j = 0; j < 4; ++j) au[j] = ktu[row + mA2 + j];
#pragma unroll
        for (int j = 0; j < 4; ++j) au[4 + j] = ktu[row + mA2 + 8 + j];
        Af[ct] = as_h16(au);
      }
#pragma unroll
      for (int dt = 0; dt < 2; ++dt) {
        const int row = (h * DH + dt * 16 + ln15) * (KT_S / 2);
        v8u bu;
#pragma unroll
        for (int j = 0; j < 8; ++j) bu[j] = vtu[row + mB2 + j + (j >> 2) * 4];
        Bf[dt] = as_h16(bu);
      }
#pragma unroll
      for (int ct = 0; ct < 2; ++ct)
#pragma unroll
        for (int dt = 0; dt < 2; ++dt)
          kv[ct][dt] = wmma_f16(Af[ct], Bf[dt], kv[ct][dt]);
    }
#pragma unroll
    for (int ct = 0; ct < 2; ++ct)
#pragma unroll
      for (int dt = 0; dt < 2; ++dt)
#pragma unroll
        for (int j = 0; j < 8; ++j) {
          const int c = ct * 16 + j + lhi * 8;
          const int d = dt * 16 + ln15;
          atomicAdd(&kv_ws[(((size_t)b * HEADS + h) * DH + c) * DH + d],
                    kv[ct][dt][j]);
        }
  }
}

// ---------------------------------------------------------------------------
// Kernel 2: q projection + output apply
// LDS: persist: kvt[8*32][34] f16 (17408B) | ksum[256] f32 | dnm[128] f32
//      union  : {xs[128][36] + wqs[256][36]}  vs  {qt[128][264] f16}
// ---------------------------------------------------------------------------
__global__ void __launch_bounds__(256)
mhla_out_kernel(const float* __restrict__ x, const float* __restrict__ wq,
                const float* __restrict__ kv_ws,
                const float* __restrict__ ksum_ws, float* __restrict__ out) {
  __shared__ __align__(16) char smem[86528];
  const int b    = blockIdx.y;
  const int m0   = blockIdx.x * MT;
  const int tid  = threadIdx.x;
  const int lane = tid & 31;
  const int wave = tid >> 5;
  const int ln15 = lane & 15;
  const int lhi  = lane >> 4;

  _Float16* kvt    = (_Float16*)smem;            // [(h,d)][c] stride KVT_S (KV^T)
  float*    ksum_s = (float*)(smem + 17408);     // [256]
  float*    dnm    = (float*)(smem + 18432);     // [128]
  char*     un     = smem + 18944;
  _Float16* xs  = (_Float16*)un;                 // [col][chan] stride XS_S
  _Float16* wqs = xs + MT * XS_S;                // [row][chan] stride W_S
  _Float16* qt  = (_Float16*)un;                 // [m][c] stride QT_S (q^T)

  // persist loads: KV^T (f16) and k_sum; float4 loads of KV
#pragma unroll
  for (int i = 0; i < (HEADS * DH * DH) / (256 * 4); ++i) {  // 8
    const int e = (tid + i * 256) * 4;
    const int h = e >> 10, c = (e >> 5) & 31, d = e & 31;
    const float4 f = *(const float4*)&kv_ws[(size_t)b * (HEADS * DH * DH) + e];
    kvt[(h * DH + d + 0) * KVT_S + c] = (_Float16)f.x;
    kvt[(h * DH + d + 1) * KVT_S + c] = (_Float16)f.y;
    kvt[(h * DH + d + 2) * KVT_S + c] = (_Float16)f.z;
    kvt[(h * DH + d + 3) * KVT_S + c] = (_Float16)f.w;
  }
  ksum_s[tid] = ksum_ws[b * COUT + tid];

  v8f acc_q[NRT];
#pragma unroll
  for (int rt = 0; rt < NRT; ++rt) acc_q[rt] = v8f_zero();

  const unsigned int* xsu  = (const unsigned int*)xs;
  const unsigned int* wqsu = (const unsigned int*)wqs;

  for (int kc = 0; kc < NKC; ++kc) {
    const int c0 = kc * KCH;
#pragma unroll
    for (int i = 0; i < (KCH * MT) / (256 * 4); ++i) {   // 4
      const int e   = (tid + i * 256) * 4;
      const int kk  = e >> 7;
      const int col = e & (MT - 1);
      const float4 f =
          *(const float4*)&x[((size_t)b * CIN + c0 + kk) * MSPAT + m0 + col];
      xs[(col + 0) * XS_S + kk] = (_Float16)f.x;
      xs[(col + 1) * XS_S + kk] = (_Float16)f.y;
      xs[(col + 2) * XS_S + kk] = (_Float16)f.z;
      xs[(col + 3) * XS_S + kk] = (_Float16)f.w;
    }
#pragma unroll
    for (int i = 0; i < (COUT * KCH) / (256 * 4); ++i) { // 8
      const int e   = (tid + i * 256) * 4;
      const int row = e >> 5;
      const int kk  = e & 31;
      const float4 fq = *(const float4*)&wq[row * CIN + c0 + kk];
      v2u pq = {pk2(fq.x, fq.y), pk2(fq.z, fq.w)};
      *(v2u*)&wqs[row * W_S + kk] = pq;
    }
    if (kc + 1 < NKC) {
      __builtin_prefetch(&x[((size_t)b * CIN + c0 + KCH) * MSPAT + m0 + lane * 4], 0, 1);
    }
    __syncthreads();
    {
      const int col = wave * 16 + ln15;
      const int kb2 = (lhi * 8) >> 1;
      v8u bu;
#pragma unroll
      for (int j = 0; j < 8; ++j) bu[j] = xsu[col * (XS_S / 2) + kb2 + j + (j >> 2) * 4];
      const v16h Bf = as_h16(bu);
      const int akb2 = (lhi * 8) >> 1;
#pragma unroll
      for (int rt = 0; rt < NRT; ++rt) {
        const int rbase = (rt * 16 + ln15) * (W_S / 2);
        v8u au;
#pragma unroll
        for (int j = 0; j < 4; ++j) au[j] = wqsu[rbase + akb2 + j];
#pragma unroll
        for (int j = 0; j < 4; ++j) au[4 + j] = wqsu[rbase + 8 + akb2 + j];
        acc_q[rt] = wmma_f16(as_h16(au), Bf, acc_q[rt]);
      }
    }
    __syncthreads();
  }

  // spill q transposed: qt[m_local][c]; one ds_store_b128 per row tile
  {
    const int mloc = wave * 16 + ln15;
#pragma unroll
    for (int rt = 0; rt < NRT; ++rt) {
      v4u qv;
#pragma unroll
      for (int jj = 0; jj < 4; ++jj)
        qv[jj] = pk2(feat_scale(acc_q[rt][2 * jj]),
                     feat_scale(acc_q[rt][2 * jj + 1]));
      // c-base = rt*16 + lhi*8 (multiple of 8); row stride 264 -> 16B aligned
      *(v4u*)&qt[mloc * QT_S + rt * 16 + lhi * 8] = qv;
    }
  }
  __syncthreads();

  // denom[m] = max(sum_c q[c,m]*ksum[c], 1e-6); packed-h2 reads
  if (tid < MT) {
    const h2* qtp = (const h2*)qt;
    float s = 0.0f;
#pragma unroll 8
    for (int c2 = 0; c2 < COUT / 2; ++c2) {
      h2 p = qtp[tid * (QT_S / 2) + c2];
      s += ksum_s[2 * c2] * (float)p[0] + ksum_s[2 * c2 + 1] * (float)p[1];
    }
    dnm[tid] = fmaxf(s, 1e-6f);
  }
  __syncthreads();

  // out[h][d][m] = (sum_c KV[c,d] q[c,m]) / denom[m]
  const unsigned int* qtu  = (const unsigned int*)qt;
  const unsigned int* kvtu = (const unsigned int*)kvt;
  const int   mloc = wave * 16 + ln15;
  const float rden = 1.0f / dnm[mloc];
  const int   akb2 = (lhi * 8) >> 1;

#pragma unroll
  for (int h = 0; h < HEADS; ++h) {
    v8u bu;
    const int cb2 = (h * DH + lhi * 8) >> 1;
#pragma unroll
    for (int j = 0; j < 8; ++j) bu[j] = qtu[mloc * (QT_S / 2) + cb2 + j + (j >> 2) * 4];
    const v16h Bf = as_h16(bu);
#pragma unroll
    for (int dt = 0; dt < 2; ++dt) {
      const int row = (h * DH + dt * 16 + ln15) * (KVT_S / 2);
      v8u au;
#pragma unroll
      for (int j = 0; j < 4; ++j) au[j] = kvtu[row + akb2 + j];
#pragma unroll
      for (int j = 0; j < 4; ++j) au[4 + j] = kvtu[row + 8 + akb2 + j];
      v8f D = wmma_f16(as_h16(au), Bf, v8f_zero());
#pragma unroll
      for (int j = 0; j < 8; ++j) {
        const int dg = dt * 16 + j + lhi * 8;
        out[((size_t)b * COUT + h * DH + dg) * MSPAT + m0 + mloc] = D[j] * rden;
      }
    }
  }
}

extern "C" void kernel_launch(void* const* d_in, const int* in_sizes, int n_in,
                              void* d_out, int out_size, void* d_ws, size_t ws_size,
                              hipStream_t stream) {
  const float* x  = (const float*)d_in[0];
  const float* wq = (const float*)d_in[1];
  const float* wk = (const float*)d_in[2];
  const float* wv = (const float*)d_in[3];
  float* out = (float*)d_out;

  float* kv_ws   = (float*)d_ws;                            // [16][8][32][32]
  float* ksum_ws = kv_ws + (size_t)BATCH * HEADS * DH * DH; // [16][256]

  const size_t ws_bytes =
      ((size_t)BATCH * HEADS * DH * DH + (size_t)BATCH * COUT) * sizeof(float);
  hipMemsetAsync(d_ws, 0, ws_bytes, stream);

  dim3 grid(MSPAT / MT, BATCH);
  dim3 block(256);
  mhla_kv_kernel<<<grid, block, 0, stream>>>(x, wk, wv, kv_ws, ksum_ws);
  mhla_out_kernel<<<grid, block, 0, stream>>>(x, wq, kv_ws, ksum_ws, out);
}
